// RBF_8289286881445
// MI455X (gfx1250) — matrix operand order repaired and checked by reading, not compiled, not run on hardware
//
#include <hip/hip_runtime.h>
#include <hip/hip_bf16.h>
#include <stdint.h>

typedef __attribute__((ext_vector_type(16))) __bf16 v16bf;
typedef __attribute__((ext_vector_type(8)))  __bf16 v8bf;
typedef __attribute__((ext_vector_type(8)))  float  v8f;

union Frag16 { v16bf v; v8bf h[2]; };

// Low 32 bits of a generic shared pointer == LDS byte address (ISA flat rules).
__device__ inline uint32_t lds_addr32(const void* p) {
    return (uint32_t)(uintptr_t)p;
}

// CDNA5 async DMA: global -> LDS, tracked by ASYNCcnt (no VGPR round-trip).
__device__ inline void async_load_b128(uint32_t lds_byte, const void* gaddr) {
    asm volatile("global_load_async_to_lds_b128 %0, %1, off"
                 :: "v"(lds_byte), "v"(gaddr)
                 : "memory");
}
__device__ inline void wait_async_zero() {
    asm volatile("s_wait_asynccnt 0x0" ::: "memory");
}

// ---------------------------------------------------------------------------
// Prepass 1: squared row norms. One wave32 per row.
// ---------------------------------------------------------------------------
__global__ __launch_bounds__(256)
void rownorm2_kernel(const float* __restrict__ m, float* __restrict__ out,
                     int rows, int cols) {
    int wave = (int)(blockIdx.x * (blockDim.x >> 5)) + (int)(threadIdx.x >> 5);
    int lane = threadIdx.x & 31;
    if (wave >= rows) return;
    const float* row = m + (size_t)wave * cols;
    float s = 0.0f;
    for (int c = lane; c < cols; c += 32) {
        float v = row[c];
        s = fmaf(v, v, s);
    }
#pragma unroll
    for (int off = 16; off > 0; off >>= 1)
        s += __shfl_down(s, off, 32);
    if (lane == 0) out[wave] = s;
}

// ---------------------------------------------------------------------------
// Prepass 2: split f32 -> bf16 hi + bf16 lo (hi+lo ~= x to ~2^-16 rel).
// ---------------------------------------------------------------------------
__global__ __launch_bounds__(256)
void convert_split_kernel(const float* __restrict__ in,
                          __bf16* __restrict__ hi, __bf16* __restrict__ lo,
                          size_t n8) {
    size_t idx = (size_t)blockIdx.x * blockDim.x + threadIdx.x;
    if (idx >= n8) return;
    size_t i = idx * 8;
    float4 a = ((const float4*)(in + i))[0];
    float4 b = ((const float4*)(in + i))[1];
    float v[8] = {a.x, a.y, a.z, a.w, b.x, b.y, b.z, b.w};
    v8bf h, l;
#pragma unroll
    for (int j = 0; j < 8; ++j) {
        __bf16 hh = (__bf16)v[j];
        h[j] = hh;
        l[j] = (__bf16)(v[j] - (float)hh);
    }
    *(v8bf*)(hi + i) = h;
    *(v8bf*)(lo + i) = l;
}

// ---------------------------------------------------------------------------
// Main GEMM with async-LDS double buffering.
// Block = 256 thr = 8 waves, block tile 128x256; wave tile 64x64 (4x4 WMMA
// tiles); 3 bf16 WMMAs per tile (hi*hi + hi*lo + lo*hi) ~ f32 accuracy.
// LDS stage: A(hi,lo) 128x32 + B(hi,lo) 256x32 bf16, row stride padded to
// 80 B (conflict-free b128 fragment reads). 2 stages = 120 KB dynamic LDS.
// ---------------------------------------------------------------------------
#define KC        32
#define RSTRIDE   80                       // padded row stride in bytes
#define OFF_AH    0
#define OFF_AL    (128 * RSTRIDE)          // 10240
#define OFF_BH    (2 * 128 * RSTRIDE)      // 20480
#define OFF_BL    (OFF_BH + 256 * RSTRIDE) // 40960
#define STAGE_SZ  (OFF_BL + 256 * RSTRIDE) // 61440
#define LDS_BYTES (2 * STAGE_SZ)           // 122880

__global__ __launch_bounds__(256)
void rbf_wmma_lds_kernel(const __bf16* __restrict__ Xh, const __bf16* __restrict__ Xl,
                         const __bf16* __restrict__ Ch, const __bf16* __restrict__ Cl,
                         const float* __restrict__ sig, const float* __restrict__ x2,
                         const float* __restrict__ c2, float* __restrict__ out,
                         int M, int N, int K) {
    extern __shared__ char smem[];
    const uint32_t smem0 = lds_addr32(smem);

    const int tid  = threadIdx.x;
    const int lane = tid & 31;
    const int w    = tid >> 5;
    const int wm   = w >> 2;              // 0..1
    const int wn   = w & 3;               // 0..3
    const int lm   = lane & 15;
    const int lh   = lane >> 4;
    const int mB   = blockIdx.x * 128;
    const int nB   = blockIdx.y * 256;

    // --- Per-thread async-copy assignments (16B chunks; 4 chunks per row) ---
    // A: 512 chunks -> 2/thread.  B: 1024 chunks -> 4/thread.
    size_t   aOff[2], bOff[4];
    uint32_t aLds[2], bLds[4];
#pragma unroll
    for (int i = 0; i < 2; ++i) {
        int c = tid + 256 * i;            // 0..511
        int row = c >> 2, kc = c & 3;
        aOff[i] = 2u * ((size_t)(mB + row) * K + kc * 8);
        aLds[i] = (uint32_t)(row * RSTRIDE + kc * 16);
    }
#pragma unroll
    for (int i = 0; i < 4; ++i) {
        int c = tid + 256 * i;            // 0..1023
        int row = c >> 2, kc = c & 3;
        bOff[i] = 2u * ((size_t)(nB + row) * K + kc * 8);
        bLds[i] = (uint32_t)(row * RSTRIDE + kc * 16);
    }

    auto issue_stage = [&](int buf, int k0) {
        uint32_t sb = smem0 + (uint32_t)(buf * STAGE_SZ);
        size_t   kb = (size_t)k0 * 2;     // byte advance along K
#pragma unroll
        for (int i = 0; i < 2; ++i) {
            async_load_b128(sb + OFF_AH + aLds[i], (const char*)Xh + aOff[i] + kb);
            async_load_b128(sb + OFF_AL + aLds[i], (const char*)Xl + aOff[i] + kb);
        }
#pragma unroll
        for (int i = 0; i < 4; ++i) {
            async_load_b128(sb + OFF_BH + bLds[i], (const char*)Ch + bOff[i] + kb);
            async_load_b128(sb + OFF_BL + bLds[i], (const char*)Cl + bOff[i] + kb);
        }
    };

    v8f acc[4][4];
#pragma unroll
    for (int mt = 0; mt < 4; ++mt)
#pragma unroll
        for (int nt = 0; nt < 4; ++nt)
#pragma unroll
            for (int r = 0; r < 8; ++r) acc[mt][nt][r] = 0.0f;

    issue_stage(0, 0);
    int cur = 0;
    for (int k0 = 0; k0 < K; k0 += KC) {
        wait_async_zero();                 // my stage-`cur` transfers done
        __syncthreads();                   // everyone's transfers done
        if (k0 + KC < K) issue_stage(cur ^ 1, k0 + KC);

        const char* s = smem + cur * STAGE_SZ;

        // B fragments (32x16): lane = column n, K chunks lh*8 and 16+lh*8.
        Frag16 bh[4], bl[4];
#pragma unroll
        for (int nt = 0; nt < 4; ++nt) {
            const char* bp = s + OFF_BH + (wn * 64 + nt * 16 + lm) * RSTRIDE + lh * 16;
            const char* lp = s + OFF_BL + (wn * 64 + nt * 16 + lm) * RSTRIDE + lh * 16;
            bh[nt].h[0] = *(const v8bf*)(bp);
            bh[nt].h[1] = *(const v8bf*)(bp + 32);
            bl[nt].h[0] = *(const v8bf*)(lp);
            bl[nt].h[1] = *(const v8bf*)(lp + 32);
        }
        // A fragments (16x32): lane = row m, K chunks lh*8 and 16+lh*8.
#pragma unroll
        for (int mt = 0; mt < 4; ++mt) {
            const char* ap = s + OFF_AH + (wm * 64 + mt * 16 + lm) * RSTRIDE + lh * 16;
            const char* lp = s + OFF_AL + (wm * 64 + mt * 16 + lm) * RSTRIDE + lh * 16;
            Frag16 ah, al;
            ah.h[0] = *(const v8bf*)(ap);
            ah.h[1] = *(const v8bf*)(ap + 32);
            al.h[0] = *(const v8bf*)(lp);
            al.h[1] = *(const v8bf*)(lp + 32);
#pragma unroll
            for (int nt = 0; nt < 4; ++nt) {
                acc[mt][nt] = __builtin_amdgcn_wmma_f32_16x16x32_bf16(
                    false, ah.v, false, bh[nt].v, (short)0, acc[mt][nt], false, false);
                acc[mt][nt] = __builtin_amdgcn_wmma_f32_16x16x32_bf16(
                    false, ah.v, false, bl[nt].v, (short)0, acc[mt][nt], false, false);
                acc[mt][nt] = __builtin_amdgcn_wmma_f32_16x16x32_bf16(
                    false, al.v, false, bh[nt].v, (short)0, acc[mt][nt], false, false);
            }
        }
        cur ^= 1;
    }

    // Fused epilogue: out = exp(-max(x2 + c2 - 2*dot, 0) * sigma^2)
    const int m0 = mB + wm * 64;
    const int n0 = nB + wn * 64;
#pragma unroll
    for (int mt = 0; mt < 4; ++mt) {
#pragma unroll
        for (int nt = 0; nt < 4; ++nt) {
            int col  = n0 + nt * 16 + lm;
            float cc = c2[col];
            float sg = sig[col];
            float s2 = sg * sg;
#pragma unroll
            for (int r = 0; r < 8; ++r) {
                int row  = m0 + mt * 16 + lh * 8 + r;
                float d2 = x2[row] + cc - 2.0f * acc[mt][nt][r];
                d2 = fmaxf(d2, 0.0f);
                out[(size_t)row * N + col] = __expf(-d2 * s2);
            }
        }
    }
}

// ---------------------------------------------------------------------------
// Fallback GEMM (fused conversion, only if workspace too small). 64x32/wave.
// ---------------------------------------------------------------------------
__device__ inline void cvt_split16(const float* __restrict__ s, v16bf& hi, v16bf& lo) {
#pragma unroll
    for (int i = 0; i < 16; ++i) {
        float x   = s[i];
        __bf16 h  = (__bf16)x;
        hi[i] = h;
        lo[i] = (__bf16)(x - (float)h);
    }
}

__global__ __launch_bounds__(256)
void rbf_wmma_fused_kernel(const float* __restrict__ X, const float* __restrict__ Cen,
                           const float* __restrict__ sig, const float* __restrict__ x2,
                           const float* __restrict__ c2, float* __restrict__ out,
                           int M, int N, int K) {
    const int lane = threadIdx.x & 31;
    const int w    = threadIdx.x >> 5;
    const int wm   = w >> 2;
    const int wn   = w & 3;
    const int m0   = blockIdx.x * 128 + wm * 64;
    const int n0   = blockIdx.y * 128 + wn * 32;
    const int lm   = lane & 15;
    const int lh   = lane >> 4;

    v8f acc[4][2];
#pragma unroll
    for (int mt = 0; mt < 4; ++mt)
#pragma unroll
        for (int nt = 0; nt < 2; ++nt)
#pragma unroll
            for (int r = 0; r < 8; ++r) acc[mt][nt][r] = 0.0f;

    for (int k0 = 0; k0 < K; k0 += 32) {
        v16bf bhi[2], blo[2];
#pragma unroll
        for (int nt = 0; nt < 2; ++nt) {
            const float* src = Cen + (size_t)(n0 + nt * 16 + lm) * K + k0 + lh * 16;
            float tmp[16];
#pragma unroll
            for (int i = 0; i < 4; ++i) {
                float4 v = ((const float4*)src)[i];
                tmp[4 * i + 0] = v.x; tmp[4 * i + 1] = v.y;
                tmp[4 * i + 2] = v.z; tmp[4 * i + 3] = v.w;
            }
            cvt_split16(tmp, bhi[nt], blo[nt]);
        }
#pragma unroll
        for (int mt = 0; mt < 4; ++mt) {
            const float* arow = X + (size_t)(m0 + mt * 16 + lm) * K + k0 + lh * 16;
            float tmp[16];
#pragma unroll
            for (int i = 0; i < 2; ++i) {
                float4 v = ((const float4*)arow)[i];
                tmp[4 * i + 0] = v.x; tmp[4 * i + 1] = v.y;
                tmp[4 * i + 2] = v.z; tmp[4 * i + 3] = v.w;
            }
#pragma unroll
            for (int i = 0; i < 2; ++i) {
                float4 v = ((const float4*)(arow + 8))[i];
                tmp[8 + 4 * i + 0] = v.x; tmp[8 + 4 * i + 1] = v.y;
                tmp[8 + 4 * i + 2] = v.z; tmp[8 + 4 * i + 3] = v.w;
            }
            v16bf ahi, alo;
            cvt_split16(tmp, ahi, alo);
#pragma unroll
            for (int nt = 0; nt < 2; ++nt) {
                acc[mt][nt] = __builtin_amdgcn_wmma_f32_16x16x32_bf16(
                    false, ahi, false, bhi[nt], (short)0, acc[mt][nt], false, false);
                acc[mt][nt] = __builtin_amdgcn_wmma_f32_16x16x32_bf16(
                    false, ahi, false, blo[nt], (short)0, acc[mt][nt], false, false);
                acc[mt][nt] = __builtin_amdgcn_wmma_f32_16x16x32_bf16(
                    false, alo, false, bhi[nt], (short)0, acc[mt][nt], false, false);
            }
        }
    }

#pragma unroll
    for (int mt = 0; mt < 4; ++mt) {
#pragma unroll
        for (int nt = 0; nt < 2; ++nt) {
            int col  = n0 + nt * 16 + lm;
            float cc = c2[col];
            float sg = sig[col];
            float s2 = sg * sg;
#pragma unroll
            for (int r = 0; r < 8; ++r) {
                int row  = m0 + mt * 16 + lh * 8 + r;
                float d2 = x2[row] + cc - 2.0f * acc[mt][nt][r];
                d2 = fmaxf(d2, 0.0f);
                out[(size_t)row * N + col] = __expf(-d2 * s2);
            }
        }
    }
}

// ---------------------------------------------------------------------------
extern "C" void kernel_launch(void* const* d_in, const int* in_sizes, int n_in,
                              void* d_out, int out_size, void* d_ws, size_t ws_size,
                              hipStream_t stream) {
    const float* x       = (const float*)d_in[0];   // [M,K]
    const float* centres = (const float*)d_in[1];   // [N,K]
    const float* sigmas  = (const float*)d_in[2];   // [N]

    const int N = in_sizes[2];                 // 2048
    const int K = in_sizes[1] / N;             // 1024
    const int M = in_sizes[0] / K;             // 8192

    float* out = (float*)d_out;
    float* x2  = (float*)d_ws;                 // M floats
    float* c2  = x2 + M;                       // N floats

    rownorm2_kernel<<<(M + 7) / 8, 256, 0, stream>>>(x, x2, M, K);
    rownorm2_kernel<<<(N + 7) / 8, 256, 0, stream>>>(centres, c2, N, K);

    const size_t normBytes = (size_t)(M + N) * sizeof(float);
    const size_t xElems    = (size_t)M * K;
    const size_t cElems    = (size_t)N * K;
    const size_t needBytes = normBytes + (xElems + cElems) * 2 * sizeof(__bf16);

    if (ws_size >= needBytes) {
        __bf16* xh = (__bf16*)((char*)d_ws + normBytes);
        __bf16* xl = xh + xElems;
        __bf16* ch = xl + xElems;
        __bf16* cl = ch + cElems;

        size_t x8 = xElems / 8, c8 = cElems / 8;
        convert_split_kernel<<<(x8 + 255) / 256, 256, 0, stream>>>(x, xh, xl, x8);
        convert_split_kernel<<<(c8 + 255) / 256, 256, 0, stream>>>(centres, ch, cl, c8);

        dim3 grid(M / 128, N / 256);
        rbf_wmma_lds_kernel<<<grid, 256, LDS_BYTES, stream>>>(xh, xl, ch, cl, sigmas,
                                                              x2, c2, out, M, N, K);
    } else {
        dim3 grid(M / 128, N / 128);
        rbf_wmma_fused_kernel<<<grid, 256, 0, stream>>>(x, centres, sigmas,
                                                        x2, c2, out, M, N, K);
    }
}
